// SlotAttention_71683004170676
// MI455X (gfx1250) — matrix operand order repaired and checked by reading, not compiled, not run on hardware
//
#include <hip/hip_runtime.h>

// ---------------- constants ----------------
constexpr int CB  = 32;     // batch
constexpr int CN  = 4096;   // tokens
constexpr int CNS = 8;      // slots
constexpr int CD  = 512;    // model dim
constexpr int MTOT = CB * CN;

typedef __attribute__((ext_vector_type(16))) __bf16 bf16x16;
typedef __attribute__((ext_vector_type(8)))  __bf16 bf16x8;
typedef __attribute__((ext_vector_type(8)))  float  f32x8;

// ---------------- WMMA helpers ----------------
static __device__ __forceinline__ f32x8 wmma_bf16(bf16x16 a, bf16x16 b, f32x8 c) {
    return __builtin_amdgcn_wmma_f32_16x16x32_bf16(false, a, false, b, (short)0, c, false, false);
}

// A fragment 16x32 bf16, source row-major [row][K] with given element stride.
// Per ISA: lanes 0-15 hold row M=lane, K={0..7,16..23}; lanes 16-31 row M=lane-16, K={8..15,24..31}.
static __device__ __forceinline__ bf16x16 frag_a(const __bf16* base, size_t row0, size_t stride, int k0) {
    const int lane = threadIdx.x & 31;
    const int m  = lane & 15;
    const int hb = (lane >> 4) * 8;
    const __bf16* p = base + (row0 + (size_t)m) * stride + (size_t)(k0 + hb);
    bf16x8 lo = *reinterpret_cast<const bf16x8*>(p);
    bf16x8 hi = *reinterpret_cast<const bf16x8*>(p + 16);
    bf16x16 r;
#pragma unroll
    for (int t = 0; t < 8; ++t) { r[t] = lo[t]; r[t + 8] = hi[t]; }
    return r;
}

// B fragment 32x16 bf16, source stored column-contiguous-in-K: base[(col)*stride + k].
// Per ISA: lanes 0-15 hold column N=lane with K={0..7,16..23}; lanes 16-31 column lane-16 with K={8..15,24..31}.
static __device__ __forceinline__ bf16x16 frag_b(const __bf16* base, size_t col0, size_t stride, int k0) {
    const int lane = threadIdx.x & 31;
    const int n  = lane & 15;
    const int hb = (lane >> 4) * 8;
    const __bf16* p = base + (col0 + (size_t)n) * stride + (size_t)(k0 + hb);
    bf16x8 lo = *reinterpret_cast<const bf16x8*>(p);
    bf16x8 hi = *reinterpret_cast<const bf16x8*>(p + 16);
    bf16x16 r;
#pragma unroll
    for (int t = 0; t < 8; ++t) { r[t] = lo[t]; r[t + 8] = hi[t]; }
    return r;
}

// C/D fragment: lanes 0-15: VGPR v -> M=v; lanes 16-31: VGPR v -> M=8+v; N = lane&15.
static __device__ __forceinline__ void store_c_f32(float* out, size_t ld, size_t col0,
                                                   f32x8 c, float scale, int rowmax) {
    const int lane = threadIdx.x & 31;
    const int n  = lane & 15;
    const int rb = (lane >> 4) << 3;
#pragma unroll
    for (int v = 0; v < 8; ++v) {
        int r = rb + v;
        if (r < rowmax) out[(size_t)r * ld + col0 + (size_t)n] = c[v] * scale;
    }
}

static __device__ __forceinline__ void store_c_bf16_bias(__bf16* out, size_t ld, size_t row0, size_t col0,
                                                         f32x8 c, const float* bias) {
    const int lane = threadIdx.x & 31;
    const int n  = lane & 15;
    const int rb = (lane >> 4) << 3;
    const float bv = bias[col0 + (size_t)n];
#pragma unroll
    for (int v = 0; v < 8; ++v) {
        out[(row0 + (size_t)(rb + v)) * ld + col0 + (size_t)n] = (__bf16)(c[v] + bv);
    }
}

// ---------------- async DMA helpers (CDNA5 ASYNCcnt path) ----------------
// LDS aperture rule (ISA 10.2): the low 32 bits of a
// generic pointer to __shared__ are the wave-relative LDS byte offset.
static __device__ __forceinline__ unsigned lds_off(const void* p) {
    return (unsigned)(uintptr_t)p;
}
// Per-lane 16B DMA: LDS[ldsOff] = MEM[gaddr]; tracked by ASYNCcnt.
static __device__ __forceinline__ void async_b128(unsigned ldsOff, const void* gp) {
    unsigned long long ga = (unsigned long long)(uintptr_t)gp;
    asm volatile("global_load_async_to_lds_b128 %0, %1, off" :: "v"(ldsOff), "v"(ga) : "memory");
}
static __device__ __forceinline__ void wait_async0() {
    asm volatile("s_wait_asynccnt 0x0" ::: "memory");
}

// ---------------- prep kernels ----------------

// cast Wk & Wv (512x512 f32, [out][in]) to bf16, layout preserved
__global__ __launch_bounds__(256) void k_cast_w(const float* __restrict__ wa, const float* __restrict__ wb,
                                                __bf16* __restrict__ oa, __bf16* __restrict__ ob, int n) {
    int i = blockIdx.x * 256 + threadIdx.x;
    if (i < n) { oa[i] = (__bf16)wa[i]; ob[i] = (__bf16)wb[i]; }
}

// LayerNorm of inputs, one wave (32 lanes) per row of 512, output bf16
__global__ __launch_bounds__(256) void k_ln_cast(const float* __restrict__ x, const float* __restrict__ g,
                                                 const float* __restrict__ be, __bf16* __restrict__ out) {
    size_t row = (size_t)blockIdx.x * 8 + (threadIdx.x >> 5);
    int lane = threadIdx.x & 31;
    const float* xr = x + row * CD;
    float vals[16];
    float s = 0.f;
#pragma unroll
    for (int t = 0; t < 16; ++t) { vals[t] = xr[lane + t * 32]; s += vals[t]; }
#pragma unroll
    for (int off = 16; off; off >>= 1) s += __shfl_xor(s, off, 32);
    float mean = s * (1.0f / CD);
    float vs = 0.f;
#pragma unroll
    for (int t = 0; t < 16; ++t) { float d = vals[t] - mean; vs += d * d; }
#pragma unroll
    for (int off = 16; off; off >>= 1) vs += __shfl_xor(vs, off, 32);
    float inv = rsqrtf(vs * (1.0f / CD) + 1e-5f);
    __bf16* orow = out + row * CD;
#pragma unroll
    for (int t = 0; t < 16; ++t) {
        int c = lane + t * 32;
        orow[c] = (__bf16)((vals[t] - mean) * inv * g[c] + be[c]);
    }
}

// slots = mu + |sigma| * noise
__global__ __launch_bounds__(256) void k_slots_init(const float* __restrict__ noise, const float* __restrict__ mu,
                                                    const float* __restrict__ sig, float* __restrict__ slots) {
    int i = blockIdx.x * 256 + threadIdx.x;   // 256*512
    int d = i & (CD - 1);
    slots[i] = mu[d] + fabsf(sig[d]) * noise[i];
}

// ---------------- big WMMA GEMM: k/v projection (async double-buffered) ----------------
// C[row, col] = sum_k A[row,k] * W[col,k] + bias[col]   (A: lnx bf16 [MTOT,512]; W bf16 [512,512])
__global__ __launch_bounds__(256) void k_gemm_kv(const __bf16* __restrict__ A,
                                                 const __bf16* __restrict__ Wk, const __bf16* __restrict__ Wv,
                                                 const float* __restrict__ bk, const float* __restrict__ bv,
                                                 __bf16* __restrict__ Ko, __bf16* __restrict__ Vo) {
    const __bf16* W   = (blockIdx.z == 0) ? Wk : Wv;
    const float* bias = (blockIdx.z == 0) ? bk : bv;
    __bf16* Out       = (blockIdx.z == 0) ? Ko : Vo;

    __shared__ __align__(16) __bf16 sA[2][128 * 40];
    __shared__ __align__(16) __bf16 sB[2][64 * 40];

    const int tid = threadIdx.x;
    const size_t row0 = (size_t)blockIdx.x * 128;
    const size_t col0 = (size_t)blockIdx.y * 64;
    const int wid = tid >> 5;
    const int wm = wid >> 1;          // 0..3
    const int wn = wid & 1;           // 0..1

    // per-thread DMA assignments (A: 128x32 tile -> 2x16B per thread; B: 64x32 -> 1x16B)
    const int rA = tid >> 1, oA = (tid & 1) * 16;
    const int cB = tid >> 2, oB = (tid & 3) * 8;
    const __bf16* gA = A + (row0 + (size_t)rA) * CD + (size_t)oA;
    const __bf16* gB = W + (col0 + (size_t)cB) * CD + (size_t)oB;
    const unsigned la0 = lds_off(&sA[0][rA * 40 + oA]);
    const unsigned la1 = lds_off(&sA[1][rA * 40 + oA]);
    const unsigned lb0 = lds_off(&sB[0][cB * 40 + oB]);
    const unsigned lb1 = lds_off(&sB[1][cB * 40 + oB]);

    // prime buffer 0
    async_b128(la0,      gA);
    async_b128(la0 + 16, gA + 8);
    async_b128(lb0,      gB);
    wait_async0();
    __syncthreads();

    f32x8 c[2][2] = {};

    for (int k0 = 0; k0 < CD; k0 += 32) {
        const int buf = (k0 >> 5) & 1;
        if (k0 + 32 < CD) {   // DMA next tile into the other buffer while we compute
            const unsigned na = buf ? la0 : la1;
            const unsigned nb = buf ? lb0 : lb1;
            async_b128(na,      gA + k0 + 32);
            async_b128(na + 16, gA + k0 + 40);
            async_b128(nb,      gB + k0 + 32);
        }
        bf16x16 a0 = frag_a(sA[buf], (size_t)(wm * 32),      40, 0);
        bf16x16 a1 = frag_a(sA[buf], (size_t)(wm * 32 + 16), 40, 0);
        bf16x16 b0 = frag_b(sB[buf], (size_t)(wn * 32),      40, 0);
        bf16x16 b1 = frag_b(sB[buf], (size_t)(wn * 32 + 16), 40, 0);
        c[0][0] = wmma_bf16(a0, b0, c[0][0]);
        c[0][1] = wmma_bf16(a0, b1, c[0][1]);
        c[1][0] = wmma_bf16(a1, b0, c[1][0]);
        c[1][1] = wmma_bf16(a1, b1, c[1][1]);
        wait_async0();
        __syncthreads();
    }
#pragma unroll
    for (int i = 0; i < 2; ++i)
#pragma unroll
        for (int j = 0; j < 2; ++j)
            store_c_bf16_bias(Out, CD, row0 + (size_t)(wm * 32 + i * 16),
                              col0 + (size_t)(wn * 32 + j * 16), c[i][j], bias);
}

// ---------------- q projection (LN + small GEMM, f32) ----------------
__global__ __launch_bounds__(256) void k_qproj(const float* __restrict__ slots, const float* __restrict__ g,
                                               const float* __restrict__ be, const float* __restrict__ Wq,
                                               const float* __restrict__ bq, __bf16* __restrict__ Qout) {
    int r = blockIdx.x;              // 0..B*16-1
    int b = r >> 4, i = r & 15;
    __bf16* orow = Qout + (size_t)r * CD;
    if (i >= CNS) {                  // zero-pad rows 8..15
        for (int c = threadIdx.x; c < CD; c += 256) orow[c] = (__bf16)0.0f;
        return;
    }
    __shared__ float y[CD];
    __shared__ float red[256];
    const float* srow = slots + ((size_t)b * CNS + i) * CD;
    float x0 = srow[threadIdx.x], x1 = srow[threadIdx.x + 256];
    red[threadIdx.x] = x0 + x1; __syncthreads();
    for (int st = 128; st; st >>= 1) { if (threadIdx.x < st) red[threadIdx.x] += red[threadIdx.x + st]; __syncthreads(); }
    float mean = red[0] * (1.0f / CD); __syncthreads();
    float d0 = x0 - mean, d1 = x1 - mean;
    red[threadIdx.x] = d0 * d0 + d1 * d1; __syncthreads();
    for (int st = 128; st; st >>= 1) { if (threadIdx.x < st) red[threadIdx.x] += red[threadIdx.x + st]; __syncthreads(); }
    float inv = rsqrtf(red[0] * (1.0f / CD) + 1e-5f); __syncthreads();
    y[threadIdx.x]       = d0 * inv * g[threadIdx.x]       + be[threadIdx.x];
    y[threadIdx.x + 256] = d1 * inv * g[threadIdx.x + 256] + be[threadIdx.x + 256];
    __syncthreads();
    for (int c = threadIdx.x; c < CD; c += 256) {
        const float* w = Wq + (size_t)c * CD;
        float acc = bq[c];
        for (int kk = 0; kk < CD; ++kk) acc += y[kk] * w[kk];
        orow[c] = (__bf16)acc;
    }
}

// ---------------- dots = q @ k^T (WMMA) ----------------
__global__ __launch_bounds__(256) void k_dots(const __bf16* __restrict__ Q,  // [B][16][512]
                                              const __bf16* __restrict__ K,  // [B*N][512]
                                              float* __restrict__ dots) {    // [B][8][N]
    const int b  = blockIdx.x;
    const int jt = blockIdx.y;       // 16 tiles of 256 columns
    __shared__ __align__(16) __bf16 sQ[16 * 520];
    const __bf16* qg = Q + (size_t)b * 16 * CD;
    for (int t = threadIdx.x; t < 16 * CD / 8; t += 256) {   // async DMA the whole q tile once
        int r = t >> 6, off = (t & 63) * 8;
        async_b128(lds_off(&sQ[r * 520 + off]), qg + (size_t)r * CD + off);
    }
    wait_async0();
    __syncthreads();
    const int wid = threadIdx.x >> 5;
    const size_t j0 = (size_t)jt * 256 + (size_t)wid * 32;
    const __bf16* kg = K + (size_t)b * CN * CD;
    f32x8 c0 = {}, c1 = {};
    for (int k0 = 0; k0 < CD; k0 += 32) {
        if (k0 + 32 < CD)   // hint the next K-slab of the streamed k^T operand
            __builtin_prefetch(kg + (j0 + (size_t)(threadIdx.x & 31)) * CD + k0 + 32, 0, 0);
        bf16x16 a  = frag_a(sQ, 0, 520, k0);
        bf16x16 b0 = frag_b(kg, j0,      CD, k0);   // k rows are contiguous-in-d = B-frag layout
        bf16x16 b1 = frag_b(kg, j0 + 16, CD, k0);
        c0 = wmma_bf16(a, b0, c0);
        c1 = wmma_bf16(a, b1, c1);
    }
    const float scale = 0.044194173824159216f;      // 512^-0.5
    float* outb = dots + (size_t)b * CNS * CN;
    store_c_f32(outb, CN, j0,      c0, scale, CNS);
    store_c_f32(outb, CN, j0 + 16, c1, scale, CNS);
}

// ---------------- softmax over slots + row sums ----------------
__global__ __launch_bounds__(256) void k_zero(float* __restrict__ p, int n) {
    int i = blockIdx.x * 256 + threadIdx.x;
    if (i < n) p[i] = 0.f;
}

__global__ __launch_bounds__(256) void k_softmax1(const float* __restrict__ dots, float* __restrict__ attn,
                                                  float* __restrict__ sums) {
    size_t idx = (size_t)blockIdx.x * 256 + threadIdx.x;   // over B*N columns
    int b = (int)(idx >> 12);
    int j = (int)(idx & (CN - 1));
    const float* dp = dots + (size_t)b * CNS * CN + j;
    float v[CNS];
    float mx = -1e30f;
#pragma unroll
    for (int i = 0; i < CNS; ++i) { v[i] = dp[(size_t)i * CN]; mx = fmaxf(mx, v[i]); }
    float se = 0.f;
#pragma unroll
    for (int i = 0; i < CNS; ++i) { v[i] = __expf(v[i] - mx); se += v[i]; }
    float inv = 1.f / se;
    float* ap = attn + (size_t)b * CNS * CN + j;
    float a[CNS];
#pragma unroll
    for (int i = 0; i < CNS; ++i) { a[i] = v[i] * inv + 1e-8f; ap[(size_t)i * CN] = a[i]; }
    __shared__ float red[256];
    for (int i = 0; i < CNS; ++i) {
        red[threadIdx.x] = a[i]; __syncthreads();
        for (int st = 128; st; st >>= 1) { if (threadIdx.x < st) red[threadIdx.x] += red[threadIdx.x + st]; __syncthreads(); }
        if (threadIdx.x == 0) atomicAdd(&sums[b * CNS + i], red[0]);
        __syncthreads();
    }
}

// normalize over tokens, cast to bf16, pad rows 8..15 with zeros
__global__ __launch_bounds__(256) void k_softmax2(const float* __restrict__ attn, const float* __restrict__ sums,
                                                  __bf16* __restrict__ attn_bf) {
    size_t idx = (size_t)blockIdx.x * 256 + threadIdx.x;   // over B*16*N
    int j = (int)(idx & (CN - 1));
    size_t t = idx >> 12;
    int i = (int)(t & 15);
    int b = (int)(t >> 4);
    __bf16 o;
    if (i < CNS) o = (__bf16)(attn[((size_t)b * CNS + i) * CN + j] / sums[b * CNS + i]);
    else         o = (__bf16)0.0f;
    attn_bf[idx] = o;
}

// ---------------- updates = attn @ v (WMMA, K=4096) ----------------
__global__ __launch_bounds__(256) void k_updates(const __bf16* __restrict__ Attn,  // [B][16][N]
                                                 const __bf16* __restrict__ V,     // [B*N][512]
                                                 float* __restrict__ Upd) {        // [B][8][512]
    const int b  = blockIdx.x;
    const int dt = blockIdx.y;       // 4 tiles of 128 columns
    __shared__ __align__(16) __bf16 sA[16 * 40];
    __shared__ __align__(16) __bf16 sV[128 * 40];   // [d_local][j] column-contiguous-K
    const __bf16* ag = Attn + (size_t)b * 16 * CN;
    const __bf16* vg = V + (size_t)b * CN * CD;
    const int d0  = dt * 128;
    const int wid = threadIdx.x >> 5;
    f32x8 c = {};
    for (int j0 = 0; j0 < CN; j0 += 32) {
        if (threadIdx.x < 64) {      // stage attn 16x32
            int r = threadIdx.x >> 2, off = (threadIdx.x & 3) * 8;
            *reinterpret_cast<bf16x8*>(&sA[r * 40 + off]) =
                *reinterpret_cast<const bf16x8*>(ag + (size_t)r * CN + j0 + off);
        }
#pragma unroll
        for (int q = 0; q < 2; ++q) {  // stage v 32j x 128d transposed
            int t  = threadIdx.x + q * 256;
            int jj = t >> 4, dc = (t & 15) * 8;
            bf16x8 vv = *reinterpret_cast<const bf16x8*>(vg + (size_t)(j0 + jj) * CD + d0 + dc);
#pragma unroll
            for (int e = 0; e < 8; ++e) sV[(dc + e) * 40 + jj] = vv[e];
        }
        __syncthreads();
        bf16x16 a  = frag_a(sA, 0, 40, 0);
        bf16x16 bb = frag_b(sV, (size_t)(wid * 16), 40, 0);
        c = wmma_bf16(a, bb, c);
        __syncthreads();
    }
    store_c_f32(Upd + (size_t)b * CNS * CD, CD, (size_t)(d0 + wid * 16), c, 1.0f, CNS);
}

// ---------------- GRU cell (f32, slot-sized) ----------------
__global__ __launch_bounds__(256) void k_gru(const float* __restrict__ upd, const float* __restrict__ slots_in,
                                             const float* __restrict__ Wih, const float* __restrict__ bih,
                                             const float* __restrict__ Whh, const float* __restrict__ bhh,
                                             float* __restrict__ slots_out) {
    int s = blockIdx.x;              // 0..255
    __shared__ float u[CD], h[CD];
    u[threadIdx.x]       = upd[(size_t)s * CD + threadIdx.x];
    u[threadIdx.x + 256] = upd[(size_t)s * CD + threadIdx.x + 256];
    h[threadIdx.x]       = slots_in[(size_t)s * CD + threadIdx.x];
    h[threadIdx.x + 256] = slots_in[(size_t)s * CD + threadIdx.x + 256];
    __syncthreads();
    for (int d = threadIdx.x; d < CD; d += 256) {
        float rx = bih[d], zx = bih[CD + d], nx = bih[2 * CD + d];
        float rh = bhh[d], zh = bhh[CD + d], nh = bhh[2 * CD + d];
        const float* w1 = Wih + (size_t)d * CD;
        const float* w2 = Wih + (size_t)(CD + d) * CD;
        const float* w3 = Wih + (size_t)(2 * CD + d) * CD;
        const float* v1 = Whh + (size_t)d * CD;
        const float* v2 = Whh + (size_t)(CD + d) * CD;
        const float* v3 = Whh + (size_t)(2 * CD + d) * CD;
        for (int kk = 0; kk < CD; ++kk) {
            float uk = u[kk], hk = h[kk];
            rx += uk * w1[kk]; zx += uk * w2[kk]; nx += uk * w3[kk];
            rh += hk * v1[kk]; zh += hk * v2[kk]; nh += hk * v3[kk];
        }
        float r = 1.f / (1.f + __expf(-(rx + rh)));
        float z = 1.f / (1.f + __expf(-(zx + zh)));
        float n = tanhf(nx + r * nh);
        slots_out[(size_t)s * CD + d] = (1.f - z) * n + z * h[d];
    }
}

// ---------------- MLP residual (f32, slot-sized) ----------------
__global__ __launch_bounds__(256) void k_mlp1(const float* __restrict__ slots, const float* __restrict__ g,
                                              const float* __restrict__ be, const float* __restrict__ W1,
                                              const float* __restrict__ b1, float* __restrict__ h1) {
    int s = blockIdx.x;
    __shared__ float y[CD];
    __shared__ float red[256];
    const float* srow = slots + (size_t)s * CD;
    float x0 = srow[threadIdx.x], x1 = srow[threadIdx.x + 256];
    red[threadIdx.x] = x0 + x1; __syncthreads();
    for (int st = 128; st; st >>= 1) { if (threadIdx.x < st) red[threadIdx.x] += red[threadIdx.x + st]; __syncthreads(); }
    float mean = red[0] * (1.0f / CD); __syncthreads();
    float d0 = x0 - mean, d1 = x1 - mean;
    red[threadIdx.x] = d0 * d0 + d1 * d1; __syncthreads();
    for (int st = 128; st; st >>= 1) { if (threadIdx.x < st) red[threadIdx.x] += red[threadIdx.x + st]; __syncthreads(); }
    float inv = rsqrtf(red[0] * (1.0f / CD) + 1e-5f); __syncthreads();
    y[threadIdx.x]       = d0 * inv * g[threadIdx.x]       + be[threadIdx.x];
    y[threadIdx.x + 256] = d1 * inv * g[threadIdx.x + 256] + be[threadIdx.x + 256];
    __syncthreads();
    for (int c = threadIdx.x; c < CD; c += 256) {
        const float* w = W1 + (size_t)c * CD;
        float acc = b1[c];
        for (int kk = 0; kk < CD; ++kk) acc += y[kk] * w[kk];
        h1[(size_t)s * CD + c] = fmaxf(acc, 0.f);
    }
}

__global__ __launch_bounds__(256) void k_mlp2(const float* __restrict__ h1, const float* __restrict__ W2,
                                              const float* __restrict__ b2, float* __restrict__ slots) {
    int s = blockIdx.x;
    __shared__ float y[CD];
    y[threadIdx.x]       = h1[(size_t)s * CD + threadIdx.x];
    y[threadIdx.x + 256] = h1[(size_t)s * CD + threadIdx.x + 256];
    __syncthreads();
    for (int d = threadIdx.x; d < CD; d += 256) {
        const float* w = W2 + (size_t)d * CD;
        float acc = b2[d];
        for (int kk = 0; kk < CD; ++kk) acc += y[kk] * w[kk];
        slots[(size_t)s * CD + d] += acc;
    }
}

__global__ __launch_bounds__(256) void k_copy(const float* __restrict__ src, float* __restrict__ dst, int n) {
    int i = blockIdx.x * 256 + threadIdx.x;
    if (i < n) dst[i] = src[i];
}

// ---------------- host launcher ----------------
extern "C" void kernel_launch(void* const* d_in, const int* in_sizes, int n_in,
                              void* d_out, int out_size, void* d_ws, size_t ws_size,
                              hipStream_t stream) {
    const float* inputs = (const float*)d_in[0];
    const float* noise  = (const float*)d_in[1];
    const float* mu     = (const float*)d_in[2];
    const float* sigma  = (const float*)d_in[3];
    const float* Wq  = (const float*)d_in[4];
    const float* bq  = (const float*)d_in[5];
    const float* Wk  = (const float*)d_in[6];
    const float* bk  = (const float*)d_in[7];
    const float* Wv  = (const float*)d_in[8];
    const float* bv  = (const float*)d_in[9];
    const float* Wih = (const float*)d_in[10];
    const float* bih = (const float*)d_in[11];
    const float* Whh = (const float*)d_in[12];
    const float* bhh = (const float*)d_in[13];
    const float* W1  = (const float*)d_in[14];
    const float* b1  = (const float*)d_in[15];
    const float* W2  = (const float*)d_in[16];
    const float* b2  = (const float*)d_in[17];
    const float* g_in  = (const float*)d_in[18];
    const float* be_in = (const float*)d_in[19];
    const float* g_sl  = (const float*)d_in[20];
    const float* be_sl = (const float*)d_in[21];
    const float* g_ff  = (const float*)d_in[22];
    const float* be_ff = (const float*)d_in[23];

    char* wp = (char*)d_ws;
    auto alloc = [&](size_t bytes) -> char* {
        char* p = wp;
        wp += (bytes + 255) & ~(size_t)255;
        return p;
    };
    __bf16* lnx    = (__bf16*)alloc((size_t)MTOT * CD * 2);
    __bf16* wk_bf  = (__bf16*)alloc((size_t)CD * CD * 2);
    __bf16* wv_bf  = (__bf16*)alloc((size_t)CD * CD * 2);
    __bf16* k_bf   = (__bf16*)alloc((size_t)MTOT * CD * 2);
    __bf16* v_bf   = (__bf16*)alloc((size_t)MTOT * CD * 2);
    __bf16* q_bf   = (__bf16*)alloc((size_t)CB * 16 * CD * 2);
    float*  dots   = (float*)alloc((size_t)CB * CNS * CN * 4);
    float*  attn   = (float*)alloc((size_t)CB * CNS * CN * 4);
    __bf16* attnbf = (__bf16*)alloc((size_t)CB * 16 * CN * 2);
    float*  sums   = (float*)alloc((size_t)CB * CNS * 4);
    float*  upd    = (float*)alloc((size_t)CB * CNS * CD * 4);
    float*  slots  = (float*)alloc((size_t)CB * CNS * CD * 4);
    float*  h1     = (float*)alloc((size_t)CB * CNS * CD * 4);

    // prep
    k_cast_w<<<(CD * CD + 255) / 256, 256, 0, stream>>>(Wk, Wv, wk_bf, wv_bf, CD * CD);
    k_ln_cast<<<MTOT / 8, 256, 0, stream>>>(inputs, g_in, be_in, lnx);
    k_slots_init<<<(CB * CNS * CD) / 256, 256, 0, stream>>>(noise, mu, sigma, slots);
    k_gemm_kv<<<dim3(MTOT / 128, CD / 64, 2), 256, 0, stream>>>(lnx, wk_bf, wv_bf, bk, bv, k_bf, v_bf);

    for (int it = 0; it < 3; ++it) {
        k_qproj<<<CB * 16, 256, 0, stream>>>(slots, g_sl, be_sl, Wq, bq, q_bf);
        k_dots<<<dim3(CB, CN / 256), 256, 0, stream>>>(q_bf, k_bf, dots);
        k_zero<<<1, 256, 0, stream>>>(sums, CB * CNS);
        k_softmax1<<<(CB * CN) / 256, 256, 0, stream>>>(dots, attn, sums);
        k_softmax2<<<(CB * 16 * CN) / 256, 256, 0, stream>>>(attn, sums, attnbf);
        k_updates<<<dim3(CB, CD / 128), 256, 0, stream>>>(attnbf, v_bf, upd);
        k_gru<<<CB * CNS, 256, 0, stream>>>(upd, slots, Wih, bih, Whh, bhh, slots);
        k_mlp1<<<CB * CNS, 256, 0, stream>>>(slots, g_ff, be_ff, W1, b1, h1);
        k_mlp2<<<CB * CNS, 256, 0, stream>>>(h1, W2, b2, slots);
    }
    k_copy<<<(CB * CNS * CD + 255) / 256, 256, 0, stream>>>(slots, (float*)d_out, CB * CNS * CD);
}
